// SpatialAttention_36558761624544
// MI455X (gfx1250) — matrix operand order repaired and checked by reading, not compiled, not run on hardware
//
#include <hip/hip_runtime.h>
#include <hip/hip_bf16.h>
#include <math.h>

#define H_DIM 1024
#define N_OBJ 64
#define BATCH 512

typedef __attribute__((ext_vector_type(16))) __bf16 v16bf;
typedef __attribute__((ext_vector_type(8)))  __bf16 v8bf;
typedef __attribute__((ext_vector_type(4)))  __bf16 v4bf;
typedef __attribute__((ext_vector_type(8)))  float  v8f;

__device__ __forceinline__ v16bf cat8(v8bf lo, v8bf hi) {
    v16bf r;
#pragma unroll
    for (int i = 0; i < 8; ++i) { r[i] = lo[i]; r[i + 8] = hi[i]; }
    return r;
}

__device__ __forceinline__ v8f wmma_bf16(v16bf a, v16bf b, v8f c) {
    // D = A(16x32 bf16) * B(32x16 bf16) + C(16x16 f32)
    return __builtin_amdgcn_wmma_f32_16x16x32_bf16(
        /*neg_a=*/false, a, /*neg_b=*/false, b,
        /*c_mod=*/(short)0, c, /*reuse_a=*/false, /*reuse_b=*/false);
}

// ---------------------------------------------------------------------------
// k0: transpose W_ua / W_wa into bf16 (K-contiguous for the WMMA B operand),
//     convert h to bf16, fold W_w's two columns into one weight vector.
// ---------------------------------------------------------------------------
__global__ void k0_prep(const float* __restrict__ Wua, const float* __restrict__ Wwa,
                        const float* __restrict__ hin, const float* __restrict__ Ww,
                        __bf16* __restrict__ WtUA, __bf16* __restrict__ WtWA,
                        __bf16* __restrict__ hbf, float* __restrict__ wsum) {
    int i = blockIdx.x * blockDim.x + threadIdx.x;
    int stride = gridDim.x * blockDim.x;
    for (int k = i; k < H_DIM * H_DIM; k += stride) {
        int r = k >> 10;            // h index (row of W)
        int c = k & (H_DIM - 1);    // k index (col of W)
        WtUA[(size_t)c * H_DIM + r] = (__bf16)Wua[k];
        WtWA[(size_t)c * H_DIM + r] = (__bf16)Wwa[k];
    }
    for (int k = i; k < BATCH * H_DIM; k += stride) hbf[k] = (__bf16)hin[k];
    for (int k = i; k < H_DIM; k += stride) wsum[k] = Ww[2 * k] + Ww[2 * k + 1];
}

// ---------------------------------------------------------------------------
// k1: dplus[b][k] = sum_h h[b][h] * W_wa[h][k] + b_a[k]   (512x1024, K=1024)
// One wave per 16x16 output tile. 4 waves / block.
// ---------------------------------------------------------------------------
__global__ void k1_dplus(const __bf16* __restrict__ hbf, const __bf16* __restrict__ WtWA,
                         const float* __restrict__ ba, float* __restrict__ dplus) {
    int lane = threadIdx.x & 31;
    int w    = threadIdx.x >> 5;
    int job  = blockIdx.x * 4 + w;          // 2048 jobs = 32 row tiles * 64 col tiles
    int nt   = job & 63;
    int R0   = (job >> 6) * 16;
    int hi   = (lane >= 16) ? 1 : 0;
    int col  = nt * 16 + (lane & 15);

    const __bf16* arow = hbf + (size_t)(R0 + (lane & 15)) * H_DIM;
    const __bf16* brow = WtWA + (size_t)col * H_DIM;

    v8f acc = {};
    for (int h0 = 0; h0 < H_DIM; h0 += 32) {
        int ka = h0 + 16 * hi;      // A: contiguous run of 16
        int kb = h0 + 16 * hi;      // B: contiguous run of 16
        v16bf a = cat8(*(const v8bf*)(arow + ka), *(const v8bf*)(arow + ka + 8));
        v16bf b = cat8(*(const v8bf*)(brow + kb), *(const v8bf*)(brow + kb + 8));
        acc = wmma_bf16(a, b, acc);
    }
    float bias = ba[col];
#pragma unroll
    for (int r = 0; r < 8; ++r) {
        int brow_i = R0 + r + 8 * hi;       // C/D: M = r + 8*hi
        dplus[(size_t)brow_i * H_DIM + col] = acc[r] + bias;
    }
}

// ---------------------------------------------------------------------------
// k2: scores[o][b] = sum_k tanh(dplus[b][k] + sum_h oe[o][b][h]*W_ua[h][k]) * w[k]
// Each wave: one 16-row (o,b) tile, N register-tiled x2 (two 16-col tiles per
// A fetch -> 2 WMMAs per pair of LDS A loads). A rows staged once in LDS as
// bf16 (32KB/wave). 4 waves/block -> 128KB dynamic LDS, 2 blocks per WGP.
// ---------------------------------------------------------------------------
__global__ void k2_scores(const float* __restrict__ obj, const __bf16* __restrict__ WtUA,
                          const float* __restrict__ dplus, const float* __restrict__ wsum,
                          float* __restrict__ scores) {
    extern __shared__ __bf16 smem[];        // 4 * 16 * 1024 bf16 = 128KB
    int lane = threadIdx.x & 31;
    int w    = threadIdx.x >> 5;
    int tile = blockIdx.x * 4 + w;          // 2048 tiles = 32768 rows / 16
    int R0   = tile * 16;
    int o    = R0 >> 9;                     // object index (tiles never cross b=512)
    int B0   = R0 & (BATCH - 1);
    __bf16* sA = smem + (size_t)w * 16 * H_DIM;

    // Stage 16 rows x 1024 of obj_embed (fp32 -> bf16) into LDS, float4-wise.
    for (int it = 0; it < 128; ++it) {
        int f   = it * 32 + lane;           // 4096 float4 units
        int row = f >> 8;
        int h4  = f & 255;
        const float4 v = *(const float4*)(obj + (((size_t)(B0 + row)) * N_OBJ + o) * H_DIM + h4 * 4);
        v4bf c;
        c[0] = (__bf16)v.x; c[1] = (__bf16)v.y; c[2] = (__bf16)v.z; c[3] = (__bf16)v.w;
        *(v4bf*)(sA + row * H_DIM + h4 * 4) = c;
    }
    __syncthreads();

    int hi = (lane >= 16) ? 1 : 0;
    const __bf16* arow = sA + (lane & 15) * H_DIM;

    float score[8];
#pragma unroll
    for (int r = 0; r < 8; ++r) score[r] = 0.f;

    for (int nt = 0; nt < 32; ++nt) {       // pairs of adjacent 16-col tiles
        int col0 = nt * 32 + (lane & 15);
        int col1 = col0 + 16;
        const __bf16* brow0 = WtUA + (size_t)col0 * H_DIM;
        const __bf16* brow1 = WtUA + (size_t)col1 * H_DIM;
        v8f acc0 = {};
        v8f acc1 = {};
        for (int h0 = 0; h0 < H_DIM; h0 += 32) {
            int ka = h0 + 16 * hi;
            int kb = h0 + 16 * hi;
            v16bf a  = cat8(*(const v8bf*)(arow + ka), *(const v8bf*)(arow + ka + 8));
            v16bf b0 = cat8(*(const v8bf*)(brow0 + kb), *(const v8bf*)(brow0 + kb + 8));
            v16bf b1 = cat8(*(const v8bf*)(brow1 + kb), *(const v8bf*)(brow1 + kb + 8));
            acc0 = wmma_bf16(a, b0, acc0);  // A reused by both WMMAs
            acc1 = wmma_bf16(a, b1, acc1);
        }
        // Fused epilogue: tanh + weighted reduction over both column tiles.
        float wk0 = wsum[col0];
        float wk1 = wsum[col1];
#pragma unroll
        for (int r = 0; r < 8; ++r) {
            int b = B0 + r + 8 * hi;
            const float* dp = dplus + (size_t)b * H_DIM;
            float e0 = tanhf(acc0[r] + dp[col0]);
            float e1 = tanhf(acc1[r] + dp[col1]);
            score[r] += e0 * wk0 + e1 * wk1;
        }
    }
    // Reduce across the 16 lanes of each half-wave (each lane covered 64 cols).
#pragma unroll
    for (int r = 0; r < 8; ++r) {
        float v = score[r];
        v += __shfl_xor(v, 8, 16);
        v += __shfl_xor(v, 4, 16);
        v += __shfl_xor(v, 2, 16);
        v += __shfl_xor(v, 1, 16);
        score[r] = v;
    }
    if ((lane & 15) == 0) {
#pragma unroll
        for (int r = 0; r < 8; ++r) {
            int b = B0 + r + 8 * hi;
            scores[(size_t)o * BATCH + b] = score[r];
        }
    }
}

// ---------------------------------------------------------------------------
// k3: softmax over objects, * zeros_object * 10, weighted sum over obj_embed.
// One block per batch element.
// ---------------------------------------------------------------------------
__global__ void k3_out(const float* __restrict__ scores, const float* __restrict__ zeros,
                       const float* __restrict__ obj, float* __restrict__ out) {
    int b = blockIdx.x;
    int tid = threadIdx.x;
    __shared__ float sv[64];
    __shared__ float red[64];

    if (tid < 64) sv[tid] = scores[tid * BATCH + b];
    __syncthreads();
    if (tid < 32) red[tid] = fmaxf(sv[tid], sv[tid + 32]);
    __syncthreads();
    if (tid < 16) red[tid] = fmaxf(red[tid], red[tid + 16]);
    __syncthreads();
    if (tid < 8)  red[tid] = fmaxf(red[tid], red[tid + 8]);
    __syncthreads();
    if (tid < 4)  red[tid] = fmaxf(red[tid], red[tid + 4]);
    __syncthreads();
    if (tid < 2)  red[tid] = fmaxf(red[tid], red[tid + 2]);
    __syncthreads();
    if (tid == 0) red[0] = fmaxf(red[0], red[1]);
    __syncthreads();
    float mx = red[0];
    __syncthreads();
    if (tid < 64) sv[tid] = expf(sv[tid] - mx);
    __syncthreads();
    if (tid < 32) red[tid] = sv[tid] + sv[tid + 32];
    __syncthreads();
    if (tid < 16) red[tid] += red[tid + 16];
    __syncthreads();
    if (tid < 8)  red[tid] += red[tid + 8];
    __syncthreads();
    if (tid < 4)  red[tid] += red[tid + 4];
    __syncthreads();
    if (tid < 2)  red[tid] += red[tid + 2];
    __syncthreads();
    if (tid == 0) red[0] += red[1];
    __syncthreads();
    float sum = red[0];
    __syncthreads();
    if (tid < 64) sv[tid] = sv[tid] / sum * zeros[tid * BATCH + b] * 10.0f;
    __syncthreads();

    // out[b][h] = sum_o alpha[o] * obj_embed[b][o][h], float4-vectorized.
    const float4* op = (const float4*)(obj + (size_t)b * N_OBJ * H_DIM);
    int h4 = tid;                        // 256 float4 per row, 256 threads
    float4 acc = make_float4(0.f, 0.f, 0.f, 0.f);
    for (int o = 0; o < N_OBJ; ++o) {
        float a = sv[o];
        float4 v = op[(size_t)o * 256 + h4];
        acc.x += a * v.x; acc.y += a * v.y; acc.z += a * v.z; acc.w += a * v.w;
    }
    ((float4*)out)[(size_t)b * 256 + h4] = acc;
}

// ---------------------------------------------------------------------------
extern "C" void kernel_launch(void* const* d_in, const int* in_sizes, int n_in,
                              void* d_out, int out_size, void* d_ws, size_t ws_size,
                              hipStream_t stream) {
    const float* obj   = (const float*)d_in[0];  // (512, 64, 1024)
    const float* hin   = (const float*)d_in[1];  // (1, 512, 1024)
    const float* zeros = (const float*)d_in[2];  // (64, 512)
    // d_in[3] = t (unused)
    const float* Ww    = (const float*)d_in[4];  // (1024, 2)
    const float* Wua   = (const float*)d_in[5];  // (1024, 1024)
    const float* ba    = (const float*)d_in[6];  // (1024,)
    const float* Wwa   = (const float*)d_in[7];  // (1024, 1024)
    float* out = (float*)d_out;                  // (512, 1, 1024)

    char* ws = (char*)d_ws;                      // ~7.2 MB used
    __bf16* WtUA  = (__bf16*)(ws);                          // 2 MB
    __bf16* WtWA  = (__bf16*)(ws + (2u << 20));             // 2 MB
    __bf16* hbf   = (__bf16*)(ws + (4u << 20));             // 1 MB
    float*  dplus = (float*) (ws + (5u << 20));             // 2 MB
    float*  wsum  = (float*) (ws + (7u << 20));             // 4 KB
    float*  scor  = (float*) (ws + (7u << 20) + 4096);      // 128 KB

    hipLaunchKernelGGL(k0_prep, dim3(1024), dim3(256), 0, stream,
                       Wua, Wwa, hin, Ww, WtUA, WtWA, hbf, wsum);
    hipLaunchKernelGGL(k1_dplus, dim3(512), dim3(128), 0, stream,
                       hbf, WtWA, ba, dplus);
    hipLaunchKernelGGL(k2_scores, dim3(512), dim3(128), 4 * 16 * H_DIM * sizeof(__bf16), stream,
                       obj, WtUA, dplus, wsum, scor);
    hipLaunchKernelGGL(k3_out, dim3(512), dim3(256), 0, stream,
                       scor, zeros, obj, out);
}
